// KAC_Attention_70360154243346
// MI455X (gfx1250) — matrix operand order repaired and checked by reading, not compiled, not run on hardware
//
#include <hip/hip_runtime.h>
#include <hip/hip_bf16.h>
#include <math.h>

// ---- problem dims ----
#define B_  8
#define C_  256
#define M_  32
#define N_  16384          // 128*128
#define EPS 1e-6f

typedef __bf16 bf16_t;
typedef __attribute__((ext_vector_type(16))) __bf16 v16bf;
typedef __attribute__((ext_vector_type(8)))  float  v8f;

union FragBF { v16bf v; uint32_t u[8]; };

// A-fragment (16x32 bf16, row-major source): lane holds pairs
// at k = 2*hi + 2j (contiguous 16 k), hi = 8 for lanes 16..31.
__device__ __forceinline__ int kpair_off(int j, int hi) {
    return 2 * hi + 2 * j;
}

__device__ __forceinline__ v16bf load_a_frag(const bf16_t* rowPtr, int hi) {
    FragBF f;
#pragma unroll
    for (int j = 0; j < 8; ++j)
        f.u[j] = *reinterpret_cast<const uint32_t*>(rowPtr + kpair_off(j, hi));
    return f.v;
}

// B-fragment (32x16 bf16): lane holds 16 contiguous K values of one column.
__device__ __forceinline__ v16bf load_b_frag(const bf16_t* p) {
    return *reinterpret_cast<const v16bf*>(p);
}

__device__ __forceinline__ float softplus_f(float z) {
    return (z > 20.0f) ? z : log1pf(expf(z));
}

#define ZERO8 {0.f,0.f,0.f,0.f,0.f,0.f,0.f,0.f}
#define WMMA_BF16(A, Bm, Cacc) \
    __builtin_amdgcn_wmma_f32_16x16x32_bf16(false, (A), false, (Bm), (short)0, (Cacc), false, false)

// ---------------- kernel 1: zero fp32 scratch ----------------
__global__ void k_zero(float* p, int count) {
    int i = blockIdx.x * blockDim.x + threadIdx.x;
    int stride = gridDim.x * blockDim.x;
    for (; i < count; i += stride) p[i] = 0.0f;
}

// ---------------- kernel 2: weights -> bf16 (W2 = [wq;wk], 64x256) ----------
__global__ void k_convw(const float* wq, const float* wk, bf16_t* W2b) {
    int idx = blockIdx.x * blockDim.x + threadIdx.x;   // < 64*256
    int row = idx >> 8, col = idx & 255;
    float v = (row < 32) ? wq[row * 256 + col] : wk[(row - 32) * 256 + col];
    W2b[idx] = (bf16_t)v;
}

// ------------- kernel 3: x -> xb (bf16 [C,N]) and xbT (bf16 [N,C]) ----------
__global__ __launch_bounds__(256) void k_xpose(const float* x, bf16_t* xb, bf16_t* xbT) {
    __shared__ float tile[32][33];
    int t = threadIdx.x;
    int lx = t & 31, ly = t >> 5;          // 32 x 8 threads
    int b  = blockIdx.z;
    int c0 = blockIdx.y * 32;
    int n0 = blockIdx.x * 32;
#pragma unroll
    for (int i = 0; i < 4; ++i) {
        int c = c0 + ly + i * 8;
        float v = x[((size_t)b * C_ + c) * N_ + n0 + lx];
        tile[ly + i * 8][lx] = v;
        xb[((size_t)b * C_ + c) * N_ + n0 + lx] = (bf16_t)v;
    }
    __syncthreads();
#pragma unroll
    for (int i = 0; i < 4; ++i) {
        int nn = n0 + ly + i * 8;
        xbT[((size_t)b * N_ + nn) * C_ + c0 + lx] = (bf16_t)tile[lx][ly + i * 8];
    }
}

// ------- kernel 4: S = softplus(W2 @ x + b) -> QTb [N,32], Kb [32,N] --------
__global__ __launch_bounds__(256) void k_qk_gemm(const bf16_t* W2b, const bf16_t* xbT,
                                                 const float* bq, const float* bk,
                                                 bf16_t* QTb, bf16_t* Kb) {
    int tid = threadIdx.x, lane = tid & 31, wave = tid >> 5;
    int w   = blockIdx.x * 8 + wave;       // 8 * 4 * 1024 waves
    int b   = w >> 12;
    int rem = w & 4095;
    int mt  = rem >> 10;                   // 0..3 (64 rows)
    int nt  = rem & 1023;                  // 0..1023
    int col = lane & 15;
    int hiA = (lane >= 16) ? 8 : 0;
    int hiB = (lane >= 16) ? 16 : 0;
    int m_row = mt * 16 + col;
    size_t n  = (size_t)nt * 16 + col;
    const bf16_t* arow = W2b + (size_t)m_row * C_;
    const bf16_t* bcol = xbT + ((size_t)b * N_ + n) * C_;
    v8f acc = ZERO8;
#pragma unroll
    for (int kt = 0; kt < 8; ++kt) {
        v16bf a  = load_a_frag(arow + kt * 32, hiA);
        v16bf bb = load_b_frag(bcol + kt * 32 + hiB);
        acc = WMMA_BF16(a, bb, acc);
    }
    int rowoff = (lane >> 4) * 8;
#pragma unroll
    for (int r = 0; r < 8; ++r) {
        int mg = mt * 16 + rowoff + r;
        float bias = (mg < 32) ? bq[mg] : bk[mg - 32];
        float v = softplus_f(acc[r] + bias);
        if (mg < 32)
            QTb[((size_t)b * N_ + n) * M_ + mg] = (bf16_t)v;          // Q^T [N,32]
        else
            Kb[((size_t)b * M_ + (mg - 32)) * N_ + n] = (bf16_t)v;    // K   [32,N]
    }
}

// ---------------- kernel 5: Ksum[b][m] = sum_n K ----------------------------
__global__ __launch_bounds__(256) void k_ksum(const bf16_t* Kb, float* Ksum) {
    int lane = threadIdx.x & 31, wave = threadIdx.x >> 5;
    int w = blockIdx.x * 8 + wave;         // 256 rows
    int b = w >> 5, m = w & 31;
    const bf16_t* row = Kb + ((size_t)b * M_ + m) * N_;
    float s = 0.0f;
    for (int i = 0; i < N_ / 32; ++i) s += (float)row[lane + i * 32];
    for (int off = 16; off; off >>= 1) s += __shfl_xor(s, off, 32);
    if (lane == 0) Ksum[b * M_ + m] = s;
}

// --- kernel 6: KX = K @ x^T  [32,256], K-dim = N, split-K, 32x32/wave -------
__global__ __launch_bounds__(256) void k_kx_gemm(const bf16_t* Kb, const bf16_t* xb,
                                                 float* KX) {
    int tid = threadIdx.x, lane = tid & 31, wave = tid >> 5;
    int w    = blockIdx.x * 8 + wave;      // 8 * 8 * 16 waves
    int b    = w >> 7;
    int rem  = w & 127;
    int nt2  = rem >> 4;                   // 0..7  (32-wide C tiles)
    int sk   = rem & 15;                   // split-K chunk
    int col  = lane & 15;
    int hiA  = (lane >= 16) ? 8 : 0;
    int hiB  = (lane >= 16) ? 16 : 0;
    const bf16_t* a0 = Kb + ((size_t)b * M_ + col) * N_;        // rows 0..15
    const bf16_t* a1 = Kb + ((size_t)b * M_ + col + 16) * N_;   // rows 16..31
    const bf16_t* b0 = xb + ((size_t)b * C_ + nt2 * 32 + col) * N_;
    const bf16_t* b1 = xb + ((size_t)b * C_ + nt2 * 32 + col + 16) * N_;
    int kstart = sk * 1024;
    v8f acc00 = ZERO8, acc01 = ZERO8, acc10 = ZERO8, acc11 = ZERO8;
    for (int kt = 0; kt < 32; ++kt) {
        int kk = kstart + kt * 32;
        __builtin_prefetch(b0 + kk + 1024, 0, 3);
        v16bf A0 = load_a_frag(a0 + kk, hiA);
        v16bf A1 = load_a_frag(a1 + kk, hiA);
        v16bf B0 = load_b_frag(b0 + kk + hiB);
        v16bf B1 = load_b_frag(b1 + kk + hiB);
        acc00 = WMMA_BF16(A0, B0, acc00);
        acc01 = WMMA_BF16(A0, B1, acc01);
        acc10 = WMMA_BF16(A1, B0, acc10);
        acc11 = WMMA_BF16(A1, B1, acc11);
    }
    int rowoff = (lane >> 4) * 8;
    int cc0 = nt2 * 32 + col, cc1 = cc0 + 16;
#pragma unroll
    for (int r = 0; r < 8; ++r) {
        int m0 = rowoff + r, m1 = m0 + 16;
        unsafeAtomicAdd(&KX[((size_t)b * M_ + m0) * C_ + cc0], acc00[r]);
        unsafeAtomicAdd(&KX[((size_t)b * M_ + m0) * C_ + cc1], acc01[r]);
        unsafeAtomicAdd(&KX[((size_t)b * M_ + m1) * C_ + cc0], acc10[r]);
        unsafeAtomicAdd(&KX[((size_t)b * M_ + m1) * C_ + cc1], acc11[r]);
    }
}

// -- kernel 7: energy = x @ x^T [256,256], K=N, split-K, 32x32 tile/wave -----
__global__ __launch_bounds__(256) void k_energy_gemm(const bf16_t* xb, float* energy) {
    int tid = threadIdx.x, lane = tid & 31, wave = tid >> 5;
    int w    = blockIdx.x * 8 + wave;      // 8 * 8 * 8 * 16 waves
    int b    = w >> 10;
    int rem  = w & 1023;
    int mt2  = rem >> 7;                   // 0..7
    int rem2 = rem & 127;
    int nt2  = rem2 >> 4;                  // 0..7
    int sk   = rem2 & 15;
    int col  = lane & 15;
    int hiA  = (lane >= 16) ? 8 : 0;
    int hiB  = (lane >= 16) ? 16 : 0;
    const bf16_t* a0 = xb + ((size_t)b * C_ + mt2 * 32 + col) * N_;
    const bf16_t* a1 = xb + ((size_t)b * C_ + mt2 * 32 + col + 16) * N_;
    const bf16_t* b0 = xb + ((size_t)b * C_ + nt2 * 32 + col) * N_;
    const bf16_t* b1 = xb + ((size_t)b * C_ + nt2 * 32 + col + 16) * N_;
    int kstart = sk * 1024;
    v8f acc00 = ZERO8, acc01 = ZERO8, acc10 = ZERO8, acc11 = ZERO8;
    for (int kt = 0; kt < 32; ++kt) {
        int kk = kstart + kt * 32;
        __builtin_prefetch(a0 + kk + 1024, 0, 3);
        __builtin_prefetch(b0 + kk + 1024, 0, 3);
        v16bf A0 = load_a_frag(a0 + kk, hiA);
        v16bf A1 = load_a_frag(a1 + kk, hiA);
        v16bf B0 = load_b_frag(b0 + kk + hiB);
        v16bf B1 = load_b_frag(b1 + kk + hiB);
        acc00 = WMMA_BF16(A0, B0, acc00);
        acc01 = WMMA_BF16(A0, B1, acc01);
        acc10 = WMMA_BF16(A1, B0, acc10);
        acc11 = WMMA_BF16(A1, B1, acc11);
    }
    int rowoff = (lane >> 4) * 8;
    int cc0 = nt2 * 32 + col, cc1 = cc0 + 16;
#pragma unroll
    for (int r = 0; r < 8; ++r) {
        int rg0 = mt2 * 32 + rowoff + r, rg1 = rg0 + 16;
        unsafeAtomicAdd(&energy[((size_t)b * C_ + rg0) * C_ + cc0], acc00[r]);
        unsafeAtomicAdd(&energy[((size_t)b * C_ + rg0) * C_ + cc1], acc01[r]);
        unsafeAtomicAdd(&energy[((size_t)b * C_ + rg1) * C_ + cc0], acc10[r]);
        unsafeAtomicAdd(&energy[((size_t)b * C_ + rg1) * C_ + cc1], acc11[r]);
    }
}

// --- kernel 8: KVT[c][m] = sum_c' KX[m][c'] wv[c][c'] + Ksum[m] bv[c] -------
__global__ void k_kv_combine(const float* KX, const float* wv, const float* bv,
                             const float* Ksum, bf16_t* KVTb) {
    int idx = blockIdx.x * blockDim.x + threadIdx.x;  // < 8*256*32
    int b   = idx >> 13;
    int rem = idx & 8191;
    int o   = rem >> 5;
    int m   = rem & 31;
    const float* kx   = KX + ((size_t)b * M_ + m) * C_;
    const float* wrow = wv + (size_t)o * C_;
    float s = Ksum[b * M_ + m] * bv[o];
    for (int c = 0; c < C_; ++c) s += kx[c] * wrow[c];
    KVTb[((size_t)b * C_ + o) * M_ + m] = (bf16_t)s;
}

// -------- kernel 9: attn = softmax(rowmax - energy) per row, -> bf16 --------
__global__ __launch_bounds__(256) void k_softmax(const float* energy, bf16_t* attnb) {
    __shared__ float red[256];
    int b = blockIdx.x >> 8, c = blockIdx.x & 255, t = threadIdx.x;
    float e = energy[((size_t)b * C_ + c) * C_ + t];
    red[t] = e; __syncthreads();
    for (int s = 128; s > 0; s >>= 1) { if (t < s) red[t] = fmaxf(red[t], red[t + s]); __syncthreads(); }
    float rowmax = red[0]; __syncthreads();
    float en = rowmax - e;
    red[t] = en; __syncthreads();
    for (int s = 128; s > 0; s >>= 1) { if (t < s) red[t] = fmaxf(red[t], red[t + s]); __syncthreads(); }
    float m2 = red[0]; __syncthreads();
    float p = expf(en - m2);
    red[t] = p; __syncthreads();
    for (int s = 128; s > 0; s >>= 1) { if (t < s) red[t] += red[t + s]; __syncthreads(); }
    float ssum = red[0];
    attnb[((size_t)b * C_ + c) * C_ + t] = (bf16_t)(p / ssum);
}

// -- kernel 10: fused epilogue (32x32 tile/wave):
//    out = x + gk*norm*(KVT@Q) + gc*(attn@x)
__global__ __launch_bounds__(256) void k_final(const float* x, const bf16_t* xbT,
                                               const bf16_t* attnb, const bf16_t* KVTb,
                                               const bf16_t* QTb, const float* Ksum,
                                               const float* gamma_k, const float* gamma_c,
                                               float* out) {
    int tid = threadIdx.x, lane = tid & 31, wave = tid >> 5;
    int w   = blockIdx.x * 8 + wave;       // 8 * 8 * 512 waves
    int b   = w >> 12;
    int rem = w & 4095;
    int ct2 = rem >> 9;                    // 0..7   (32-wide C tiles)
    int nt2 = rem & 511;                   // 0..511 (32-wide N tiles)
    int col = lane & 15;
    int hiA = (lane >= 16) ? 8 : 0;
    int hiB = (lane >= 16) ? 16 : 0;
    int c0  = ct2 * 32 + col, c1 = c0 + 16;
    size_t n0 = (size_t)nt2 * 32 + col, n1 = n0 + 16;

    // cam = attn @ x  (K = C = 256)
    const bf16_t* ar0 = attnb + ((size_t)b * C_ + c0) * C_;
    const bf16_t* ar1 = attnb + ((size_t)b * C_ + c1) * C_;
    const bf16_t* br0 = xbT + ((size_t)b * N_ + n0) * C_;
    const bf16_t* br1 = xbT + ((size_t)b * N_ + n1) * C_;
    v8f cC00 = ZERO8, cC01 = ZERO8, cC10 = ZERO8, cC11 = ZERO8;
#pragma unroll
    for (int kt = 0; kt < 8; ++kt) {
        v16bf A0 = load_a_frag(ar0 + kt * 32, hiA);
        v16bf A1 = load_a_frag(ar1 + kt * 32, hiA);
        v16bf B0 = load_b_frag(br0 + kt * 32 + hiB);
        v16bf B1 = load_b_frag(br1 + kt * 32 + hiB);
        cC00 = WMMA_BF16(A0, B0, cC00);
        cC01 = WMMA_BF16(A0, B1, cC01);
        cC10 = WMMA_BF16(A1, B0, cC10);
        cC11 = WMMA_BF16(A1, B1, cC11);
    }
    // P = KV^T @ Q  (K = M = 32 -> one WMMA per sub-tile)
    v8f cP00 = ZERO8, cP01 = ZERO8, cP10 = ZERO8, cP11 = ZERO8;
    {
        v16bf A0 = load_a_frag(KVTb + ((size_t)b * C_ + c0) * M_, hiA);
        v16bf A1 = load_a_frag(KVTb + ((size_t)b * C_ + c1) * M_, hiA);
        v16bf B0 = load_b_frag(QTb + ((size_t)b * N_ + n0) * M_ + hiB);
        v16bf B1 = load_b_frag(QTb + ((size_t)b * N_ + n1) * M_ + hiB);
        cP00 = WMMA_BF16(A0, B0, cP00);
        cP01 = WMMA_BF16(A0, B1, cP01);
        cP10 = WMMA_BF16(A1, B0, cP10);
        cP11 = WMMA_BF16(A1, B1, cP11);
    }
    // norm[n] = 1 / (Q[n,:] . (Ksum + eps)) for both columns held by this lane
    const bf16_t* q0 = QTb + ((size_t)b * N_ + n0) * M_;
    const bf16_t* q1 = QTb + ((size_t)b * N_ + n1) * M_;
    const float*  ks = Ksum + b * M_;
    float s0 = 0.0f, s1 = 0.0f;
#pragma unroll
    for (int m = 0; m < M_; ++m) {
        float kse = ks[m] + EPS;
        s0 += (float)q0[m] * kse;
        s1 += (float)q1[m] * kse;
    }
    float nrm0 = 1.0f / s0, nrm1 = 1.0f / s1;
    float gk = gamma_k[0], gc = gamma_c[0];

    int rowoff = (lane >> 4) * 8;
#pragma unroll
    for (int r = 0; r < 8; ++r) {
        int cr0 = ct2 * 32 + rowoff + r, cr1 = cr0 + 16;
        size_t i00 = ((size_t)b * C_ + cr0) * N_ + n0;
        size_t i01 = ((size_t)b * C_ + cr0) * N_ + n1;
        size_t i10 = ((size_t)b * C_ + cr1) * N_ + n0;
        size_t i11 = ((size_t)b * C_ + cr1) * N_ + n1;
        out[i00] = x[i00] + gk * nrm0 * cP00[r] + gc * cC00[r];
        out[i01] = x[i01] + gk * nrm1 * cP01[r] + gc * cC01[r];
        out[i10] = x[i10] + gk * nrm0 * cP10[r] + gc * cC10[r];
        out[i11] = x[i11] + gk * nrm1 * cP11[r] + gc * cC11[r];
    }
}

// ---------------- launcher ----------------
extern "C" void kernel_launch(void* const* d_in, const int* in_sizes, int n_in,
                              void* d_out, int out_size, void* d_ws, size_t ws_size,
                              hipStream_t stream) {
    const float* x   = (const float*)d_in[0];
    const float* wq  = (const float*)d_in[1];
    const float* bq  = (const float*)d_in[2];
    const float* wk  = (const float*)d_in[3];
    const float* bk  = (const float*)d_in[4];
    const float* wv  = (const float*)d_in[5];
    const float* bv  = (const float*)d_in[6];
    const float* gk  = (const float*)d_in[7];
    const float* gc  = (const float*)d_in[8];
    float* out       = (float*)d_out;

    char* ws = (char*)d_ws;
    const size_t XB_OFF   = 0;                                  // 64 MB bf16 [B,C,N]
    const size_t XBT_OFF  = XB_OFF  + (size_t)B_ * C_ * N_ * 2; // 64 MB bf16 [B,N,C]
    const size_t W2_OFF   = XBT_OFF + (size_t)B_ * C_ * N_ * 2; // 32 KB
    const size_t KB_OFF   = W2_OFF  + (size_t)64 * C_ * 2;      // 8 MB  bf16 [B,32,N]
    const size_t QT_OFF   = KB_OFF  + (size_t)B_ * M_ * N_ * 2; // 8 MB  bf16 [B,N,32]
    const size_t KX_OFF   = QT_OFF  + (size_t)B_ * N_ * M_ * 2; // 256 KB f32 [B,32,C]
    const size_t KSUM_OFF = KX_OFF  + (size_t)B_ * M_ * C_ * 4; // 1 KB  f32 [B,32]
    const size_t KVT_OFF  = KSUM_OFF+ (size_t)B_ * M_ * 4;      // 128 KB bf16 [B,C,32]
    const size_t EN_OFF   = KVT_OFF + (size_t)B_ * C_ * M_ * 2; // 2 MB  f32 [B,C,C]
    const size_t AT_OFF   = EN_OFF  + (size_t)B_ * C_ * C_ * 4; // 1 MB  bf16 [B,C,C]

    bf16_t* xb    = (bf16_t*)(ws + XB_OFF);
    bf16_t* xbT   = (bf16_t*)(ws + XBT_OFF);
    bf16_t* W2b   = (bf16_t*)(ws + W2_OFF);
    bf16_t* Kb    = (bf16_t*)(ws + KB_OFF);
    bf16_t* QTb   = (bf16_t*)(ws + QT_OFF);
    float*  KX    = (float*)(ws + KX_OFF);
    float*  Ksum  = (float*)(ws + KSUM_OFF);
    bf16_t* KVTb  = (bf16_t*)(ws + KVT_OFF);
    float*  energy= (float*)(ws + EN_OFF);
    bf16_t* attnb = (bf16_t*)(ws + AT_OFF);

    // 1) zero atomic accumulators
    k_zero<<<256,  256, 0, stream>>>(KX, B_ * M_ * C_);
    k_zero<<<2048, 256, 0, stream>>>(energy, B_ * C_ * C_);
    // 2) weights -> bf16
    k_convw<<<64, 256, 0, stream>>>(wq, wk, W2b);
    // 3) x -> xb, xbT
    k_xpose<<<dim3(N_ / 32, C_ / 32, B_), 256, 0, stream>>>(x, xb, xbT);
    // 4) Q/K projections + softplus
    k_qk_gemm<<<4096, 256, 0, stream>>>(W2b, xbT, bq, bk, QTb, Kb);
    // 5) Ksum
    k_ksum<<<32, 256, 0, stream>>>(Kb, Ksum);
    // 6) KX = K @ x^T (split-K atomics, 32x32/wave)
    k_kx_gemm<<<128, 256, 0, stream>>>(Kb, xb, KX);
    // 7) energy = x @ x^T (split-K atomics, 32x32/wave)
    k_energy_gemm<<<1024, 256, 0, stream>>>(xb, energy);
    // 8) KV^T = KX @ wv^T + Ksum (x) bv
    k_kv_combine<<<256, 256, 0, stream>>>(KX, wv, bv, Ksum, KVTb);
    // 9) channel softmax
    k_softmax<<<B_ * C_, 256, 0, stream>>>(energy, attnb);
    // 10) fused output (32x32/wave)
    k_final<<<4096, 256, 0, stream>>>(x, xbT, attnb, KVTb, QTb, Ksum, gk, gc, out);
}
